// RAILTEncoder_59889023975799
// MI455X (gfx1250) — matrix operand order repaired and checked by reading, not compiled, not run on hardware
//
#include <hip/hip_runtime.h>

// ---------------------------------------------------------------------------
// Types for CDNA5 WMMA (gfx1250, wave32)
// ---------------------------------------------------------------------------
typedef __attribute__((ext_vector_type(16))) __bf16       bf16x16;
typedef __attribute__((ext_vector_type(8)))  float        floatx8;
typedef __attribute__((ext_vector_type(4)))  float        floatx4;
typedef __attribute__((ext_vector_type(4)))  unsigned int uintx4;

union FragBF {
    bf16x16 bf;
    uintx4  u[2];
};

// ---------------------------------------------------------------------------
// Problem constants (from reference)
// ---------------------------------------------------------------------------
constexpr int kB     = 32;
constexpr int kL     = 2048;
constexpr int kD     = 1024;
constexpr int kH     = 16;
constexpr int kChunk = 128;
constexpr int kMem   = 64;
constexpr int kT     = kMem + kChunk;   // 192
constexpr int kHD    = kD / kH;         // 64
constexpr int kNC    = kL / kChunk;     // 16
constexpr int kD3    = 3 * kD;          // 3072
constexpr int kD4    = 4 * kD;          // 4096

__device__ inline unsigned short f2bf(float f) {
    unsigned int u = __float_as_uint(f);
    u += 0x7FFFu + ((u >> 16) & 1u);    // round-to-nearest-even
    return (unsigned short)(u >> 16);
}

// ---------------------------------------------------------------------------
// CDNA5 async global->LDS copy (ASYNCcnt) helpers
// ---------------------------------------------------------------------------
__device__ inline void async_copy_b128(unsigned ldsAddr,
                                       const unsigned short* gaddr) {
    asm volatile("global_load_async_to_lds_b128 %0, %1, off"
                 :: "v"(ldsAddr), "v"(gaddr)
                 : "memory");
}
__device__ inline void wait_async_all() {
    asm volatile("s_wait_asynccnt 0x0" ::: "memory");
}

// ---------------------------------------------------------------------------
// fp32 -> bf16 weight conversion (grid-stride)
// ---------------------------------------------------------------------------
__global__ void cvt_bf16_kernel(const float* __restrict__ src,
                                unsigned short* __restrict__ dst, int n) {
    for (int i = blockIdx.x * blockDim.x + threadIdx.x; i < n;
         i += gridDim.x * blockDim.x)
        dst[i] = f2bf(src[i]);
}

// ---------------------------------------------------------------------------
// Block-wide LayerNorm over one row of D=1024 (256 thr x 4 elems)
// ---------------------------------------------------------------------------
__device__ inline void layernorm_row_store(floatx4 zv, int row, int d0,
                                           const float* __restrict__ g,
                                           const float* __restrict__ bta,
                                           unsigned short* __restrict__ out) {
    __shared__ float part[16];
    __shared__ float sstat[2];
    const int wave = threadIdx.x >> 5;
    const int lane = threadIdx.x & 31;
    float s  = zv.x + zv.y + zv.z + zv.w;
    float ss = zv.x * zv.x + zv.y * zv.y + zv.z * zv.z + zv.w * zv.w;
#pragma unroll
    for (int o = 16; o > 0; o >>= 1) {
        s  += __shfl_down(s, o);
        ss += __shfl_down(ss, o);
    }
    if (lane == 0) { part[wave] = s; part[8 + wave] = ss; }
    __syncthreads();
    if (threadIdx.x == 0) {
        float a = 0.f, q = 0.f;
#pragma unroll
        for (int i = 0; i < 8; ++i) { a += part[i]; q += part[8 + i]; }
        float mu  = a * (1.f / (float)kD);
        float var = q * (1.f / (float)kD) - mu * mu;
        sstat[0]  = mu;
        sstat[1]  = rsqrtf(var + 1e-5f);
    }
    __syncthreads();
    const float mu = sstat[0], rs = sstat[1];
    unsigned short* o4 = out + (size_t)row * kD + d0;
    o4[0] = f2bf((zv.x - mu) * rs * g[d0 + 0] + bta[d0 + 0]);
    o4[1] = f2bf((zv.y - mu) * rs * g[d0 + 1] + bta[d0 + 1]);
    o4[2] = f2bf((zv.z - mu) * rs * g[d0 + 2] + bta[d0 + 2]);
    o4[3] = f2bf((zv.w - mu) * rs * g[d0 + 3] + bta[d0 + 3]);
}

// ---------------------------------------------------------------------------
// z = concat(mem, emb_gather) + pos  (f32), zln = LN(z) (bf16)
// grid = B*T rows, 256 threads
// ---------------------------------------------------------------------------
__global__ __launch_bounds__(256)
void build_z_ln_kernel(const int* __restrict__ in_prob,
                       const int* __restrict__ in_skill,
                       const float* __restrict__ emb_prob,
                       const float* __restrict__ emb_skill,
                       const float* __restrict__ mem_prev,
                       const float* __restrict__ mem_init,
                       const float* __restrict__ pos,
                       const float* __restrict__ lng,
                       const float* __restrict__ lnb,
                       float* __restrict__ z,
                       unsigned short* __restrict__ zln, int step) {
    const int row = blockIdx.x;
    const int b = row / kT, t = row % kT;
    const int d0 = threadIdx.x * 4;
    floatx4 base;
    if (t < kMem) {
        const float* src = (step == 0)
                               ? (mem_init + (size_t)t * kD)
                               : (mem_prev + ((size_t)b * kMem + t) * kD);
        base = *(const floatx4*)(src + d0);
    } else {
        const int li = step * kChunk + (t - kMem);
        const int ip = in_prob[b * kL + li];
        const int is = in_skill[b * kL + li];
        floatx4 e1 = *(const floatx4*)(emb_prob + (size_t)ip * kD + d0);
        floatx4 e2 = *(const floatx4*)(emb_skill + (size_t)is * kD + d0);
        base = e1 + e2;
    }
    floatx4 pv = *(const floatx4*)(pos + (size_t)t * kD + d0);
    floatx4 zv = base + pv;
    *(floatx4*)(z + (size_t)row * kD + d0) = zv;
    layernorm_row_store(zv, row, d0, lng, lnb, zln);
}

// ---------------------------------------------------------------------------
// zln = LN(z) (bf16) for the FFN branch
// ---------------------------------------------------------------------------
__global__ __launch_bounds__(256)
void ln_kernel(const float* __restrict__ z, const float* __restrict__ g,
               const float* __restrict__ b, unsigned short* __restrict__ out) {
    const int row = blockIdx.x;
    const int d0 = threadIdx.x * 4;
    floatx4 zv = *(const floatx4*)(z + (size_t)row * kD + d0);
    layernorm_row_store(zv, row, d0, g, b, out);
}

// ---------------------------------------------------------------------------
// Generic bf16 WMMA GEMM:  C[M,N] = A[M,K] * W[N,K]^T + bias
// Block = 256 threads = 8 waves arranged 4(m) x 2(n).
// Each wave computes a 2x2 grid of 16x16 tiles (32x32 C) -> block tile 128x64.
// A tile (128x32 per k-step) staged into LDS with async global->LDS copies,
// double-buffered; B (weights) fragments loaded directly (K-contiguous, L2
// resident across steps).
// MODE 0: out bf16              (QKV)
// MODE 1: outF[idx] += r        (residual: attn proj / FFN2)
// MODE 2: out bf16 = relu(r)    (FFN1)
// MODE 3: mem blend: g=sigmoid(r); memNew = g*cand + (1-g)*memOld
// ---------------------------------------------------------------------------
__device__ inline void stage_a_tile(const unsigned short* __restrict__ A,
                                    unsigned ldsBase, int buf, size_t mBlock,
                                    int K, int k) {
    // 128 rows x 32 bf16 = 512 x 16B chunks; 2 chunks per thread.
    const int c0 = threadIdx.x * 2;
#pragma unroll
    for (int cc = 0; cc < 2; ++cc) {
        const int c = c0 + cc;
        const int r = c >> 2;
        const int part = c & 3;
        const unsigned lds =
            ldsBase + (unsigned)(((buf * 128 + r) * 32 + part * 8) * 2);
        const unsigned short* g = A + (mBlock + (size_t)r) * K + k + part * 8;
        async_copy_b128(lds, g);
    }
}

template <int MODE>
__global__ __launch_bounds__(256)
void wmma_gemm_kernel(const unsigned short* __restrict__ A,
                      const unsigned short* __restrict__ W,
                      const float* __restrict__ bias,
                      unsigned short* __restrict__ outBf,
                      float* __restrict__ outF,          // MODE1 dst / MODE3 z (cand)
                      float* __restrict__ memNew,        // MODE3
                      const float* __restrict__ memOld,  // MODE3
                      int M, int N, int K) {
    __shared__ __align__(16) unsigned short ldsA[2][128][32];  // 16KB

    const int wave  = threadIdx.x >> 5;
    const int lane  = threadIdx.x & 31;
    const int half  = lane >> 4;
    const int l15   = lane & 15;
    const int waveM = wave >> 1;  // 0..3
    const int waveN = wave & 1;   // 0..1
    const size_t mBlock = (size_t)blockIdx.y * 128;
    const int    nWave  = blockIdx.x * 64 + waveN * 32;

    const unsigned ldsBase = (unsigned)(uintptr_t)(&ldsA[0][0][0]);

    // B frag rows (weights, [N,K] row-major: K pairs split at lane-half)
    const unsigned short* brow0 = W + (size_t)(nWave + l15) * K + half * 8;
    const unsigned short* brow1 = W + (size_t)(nWave + 16 + l15) * K + half * 8;

    floatx8 acc[2][2];
    const floatx8 zero8 = {0.f, 0.f, 0.f, 0.f, 0.f, 0.f, 0.f, 0.f};
#pragma unroll
    for (int ti = 0; ti < 2; ++ti)
#pragma unroll
        for (int tj = 0; tj < 2; ++tj) acc[ti][tj] = zero8;

    // prologue: stage k=0 A tile
    stage_a_tile(A, ldsBase, 0, mBlock, K, 0);
    wait_async_all();
    __syncthreads();

    for (int k = 0; k < K; k += 32) {
        const int buf = (k >> 5) & 1;
        const bool more = (k + 32) < K;
        if (more) stage_a_tile(A, ldsBase, buf ^ 1, mBlock, K, k + 32);

        // A fragments from LDS (lane = row, K pairs split at lane-half)
        FragBF a[2];
#pragma unroll
        for (int ti = 0; ti < 2; ++ti) {
            const unsigned short* lrow =
                &ldsA[buf][waveM * 32 + ti * 16 + l15][half * 8];
            a[ti].u[0] = *(const uintx4*)(lrow);
            a[ti].u[1] = *(const uintx4*)(lrow + 16);
        }
        // B fragments direct from global (L2-resident weights)
        FragBF b[2];
        b[0].u[0] = *(const uintx4*)(brow0 + k);
        b[0].u[1] = *(const uintx4*)(brow0 + k + 16);
        b[1].u[0] = *(const uintx4*)(brow1 + k);
        b[1].u[1] = *(const uintx4*)(brow1 + k + 16);
        __builtin_prefetch(brow0 + k + 32, 0, 1);
        __builtin_prefetch(brow1 + k + 32, 0, 1);

#pragma unroll
        for (int ti = 0; ti < 2; ++ti)
#pragma unroll
            for (int tj = 0; tj < 2; ++tj)
                acc[ti][tj] = __builtin_amdgcn_wmma_f32_16x16x32_bf16(
                    false, a[ti].bf, false, b[tj].bf, (short)0, acc[ti][tj],
                    false, false);

        if (more) {
            wait_async_all();   // copy for k+32 done (overlapped with WMMAs)
            __syncthreads();    // visible to all waves; safe to reuse buffers
        }
    }

#pragma unroll
    for (int tj = 0; tj < 2; ++tj) {
        const int col = nWave + tj * 16 + l15;
        const float bv = bias[col];
#pragma unroll
        for (int ti = 0; ti < 2; ++ti) {
#pragma unroll
            for (int v = 0; v < 8; ++v) {
                const int row =
                    (int)mBlock + waveM * 32 + ti * 16 + half * 8 + v;
                const size_t idx = (size_t)row * N + col;
                const float r = acc[ti][tj][v] + bv;
                if (MODE == 0) {
                    outBf[idx] = f2bf(r);
                } else if (MODE == 1) {
                    outF[idx] += r;
                } else if (MODE == 2) {
                    outBf[idx] = f2bf(fmaxf(r, 0.f));
                } else {  // MODE 3: gated memory update; row = b*kMem + t
                    const float gv = 1.f / (1.f + __expf(-r));
                    const size_t zi =
                        ((size_t)((row >> 6) * kT + (row & 63))) * kD + col;
                    const float c = outF[zi];  // cand (= z[:, :MEM])
                    memNew[idx] = gv * c + (1.f - gv) * memOld[idx];
                }
            }
        }
    }
}

// ---------------------------------------------------------------------------
// Attention: one block per (b,h). 192 threads = 6 waves, 2 Q-tiles per wave.
// Q*K^T via WMMA (frags from global bf16 qkv), masked softmax with 16-lane
// shfl reductions, P staged per-wave in LDS, P*V via WMMA (V^T in LDS).
// ---------------------------------------------------------------------------
constexpr int kAttWaves = 6;

__global__ __launch_bounds__(192)
void attn_kernel(const unsigned short* __restrict__ qkv,
                 unsigned short* __restrict__ o) {
    __shared__ __align__(16) unsigned short vt[kHD * kT];               // 24KB
    __shared__ __align__(16) unsigned short pbuf[kAttWaves * 16 * kT];  // 36KB

    const int b = blockIdx.x / kH;
    const int h = blockIdx.x % kH;
    const unsigned short* base = qkv + (size_t)b * kT * kD3;

    // stage V transposed: vt[d * kT + t]
    for (int i = threadIdx.x; i < kHD * kT; i += 192) {
        const int d = i / kT, t = i % kT;
        vt[i] = base[(size_t)t * kD3 + 2 * kD + h * kHD + d];
    }
    __syncthreads();

    const int wave = threadIdx.x >> 5;
    const int lane = threadIdx.x & 31;
    const int half = lane >> 4;
    const int l15  = lane & 15;
    unsigned short* pw = pbuf + wave * 16 * kT;

    for (int tile = wave; tile < kT / 16; tile += kAttWaves) {
        const int m0 = tile * 16;

        // Q fragments (16 x 64, two k-steps of 32)
        FragBF qa[2];
        {
            const unsigned short* qrow =
                base + (size_t)(m0 + l15) * kD3 + h * kHD + half * 8;
#pragma unroll
            for (int ks = 0; ks < 2; ++ks) {
                qa[ks].u[0] = *(const uintx4*)(qrow + ks * 32);
                qa[ks].u[1] = *(const uintx4*)(qrow + ks * 32 + 16);
            }
        }

        floatx8 S[12];
        const floatx8 zero8 = {0.f, 0.f, 0.f, 0.f, 0.f, 0.f, 0.f, 0.f};
#pragma unroll
        for (int nt = 0; nt < 12; ++nt) S[nt] = zero8;

        // S = Q * K^T
#pragma unroll
        for (int nt = 0; nt < 12; ++nt) {
            const unsigned short* krow =
                base + (size_t)(nt * 16 + l15) * kD3 + kD + h * kHD + half * 8;
#pragma unroll
            for (int ks = 0; ks < 2; ++ks) {
                FragBF kb;
                kb.u[0] = *(const uintx4*)(krow + ks * 32);
                kb.u[1] = *(const uintx4*)(krow + ks * 32 + 16);
                S[nt] = __builtin_amdgcn_wmma_f32_16x16x32_bf16(
                    false, qa[ks].bf, false, kb.bf, (short)0, S[nt], false,
                    false);
            }
        }

        // masked softmax over 192 columns per row; write P (bf16) to LDS
#pragma unroll
        for (int v = 0; v < 8; ++v) {
            const int qg = m0 + half * 8 + v;
            float mx = -3.0e38f;
#pragma unroll
            for (int nt = 0; nt < 12; ++nt) {
                const int kg = nt * 16 + l15;
                float sv = S[nt][v] * 0.125f;  // 1/sqrt(64)
                if (qg >= kMem && kg >= kMem && kg > qg) sv = -3.0e38f;
                S[nt][v] = sv;
                mx = fmaxf(mx, sv);
            }
#pragma unroll
            for (int msk = 8; msk > 0; msk >>= 1)
                mx = fmaxf(mx, __shfl_xor(mx, msk));
            float sum = 0.f;
#pragma unroll
            for (int nt = 0; nt < 12; ++nt) {
                const float e = __expf(S[nt][v] - mx);
                S[nt][v] = e;
                sum += e;
            }
#pragma unroll
            for (int msk = 8; msk > 0; msk >>= 1) sum += __shfl_xor(sum, msk);
            const float inv = 1.f / sum;
#pragma unroll
            for (int nt = 0; nt < 12; ++nt)
                pw[(half * 8 + v) * kT + nt * 16 + l15] =
                    f2bf(S[nt][v] * inv);
        }

        // O = P * V  (K = 192, 6 k-steps; 4 output tiles of 16 cols)
        floatx8 Oacc[4];
#pragma unroll
        for (int n0 = 0; n0 < 4; ++n0) Oacc[n0] = zero8;
#pragma unroll
        for (int kt = 0; kt < 6; ++kt) {
            FragBF pa;
            const unsigned short* prow = pw + l15 * kT + kt * 32 + half * 8;
            pa.u[0] = *(const uintx4*)(prow);
            pa.u[1] = *(const uintx4*)(prow + 16);
#pragma unroll
            for (int n0 = 0; n0 < 4; ++n0) {
                FragBF vb;
                const unsigned short* vrow =
                    vt + (n0 * 16 + l15) * kT + kt * 32 + half * 8;
                vb.u[0] = *(const uintx4*)(vrow);
                vb.u[1] = *(const uintx4*)(vrow + 16);
                Oacc[n0] = __builtin_amdgcn_wmma_f32_16x16x32_bf16(
                    false, pa.bf, false, vb.bf, (short)0, Oacc[n0], false,
                    false);
            }
        }
#pragma unroll
        for (int n0 = 0; n0 < 4; ++n0)
#pragma unroll
            for (int v = 0; v < 8; ++v)
                o[((size_t)b * kT + m0 + half * 8 + v) * kD + h * kHD +
                  n0 * 16 + l15] = f2bf(Oacc[n0][v]);
    }
}

// ---------------------------------------------------------------------------
// Write h-slice of z to output; convert cand slice to bf16 for gate GEMM
// ---------------------------------------------------------------------------
__global__ __launch_bounds__(256)
void finalize_kernel(const float* __restrict__ z, float* __restrict__ out,
                     unsigned short* __restrict__ candBf, int step) {
    const int row = blockIdx.x;
    const int b = row / kT, t = row % kT;
    const int d0 = threadIdx.x * 4;
    floatx4 zv = *(const floatx4*)(z + (size_t)row * kD + d0);
    if (t >= kMem) {
        *(floatx4*)(out +
                    ((size_t)b * kL + step * kChunk + (t - kMem)) * kD + d0) =
            zv;
    } else {
        unsigned short* c = candBf + ((size_t)b * kMem + t) * kD + d0;
        c[0] = f2bf(zv.x);
        c[1] = f2bf(zv.y);
        c[2] = f2bf(zv.z);
        c[3] = f2bf(zv.w);
    }
}

// step 0: mem_new = cand (no gate)
__global__ __launch_bounds__(256)
void copy_cand_kernel(const float* __restrict__ z, float* __restrict__ memNew) {
    const int row = blockIdx.x;  // b*kMem + t
    const int b = row / kMem, t = row % kMem;
    const int d0 = threadIdx.x * 4;
    *(floatx4*)(memNew + (size_t)row * kD + d0) =
        *(const floatx4*)(z + ((size_t)b * kT + t) * kD + d0);
}

// ---------------------------------------------------------------------------
// Host orchestration
// ---------------------------------------------------------------------------
extern "C" void kernel_launch(void* const* d_in, const int* in_sizes, int n_in,
                              void* d_out, int out_size, void* d_ws,
                              size_t ws_size, hipStream_t stream) {
    const int*   in_prob   = (const int*)d_in[0];
    const int*   in_skill  = (const int*)d_in[1];
    const float* emb_prob  = (const float*)d_in[2];
    const float* emb_skill = (const float*)d_in[3];
    const float* mem_init  = (const float*)d_in[4];
    const float* pos_emb   = (const float*)d_in[5];
    const float* wqkv      = (const float*)d_in[6];
    const float* bqkv      = (const float*)d_in[7];
    const float* wo        = (const float*)d_in[8];
    const float* bo        = (const float*)d_in[9];
    const float* w1        = (const float*)d_in[10];
    const float* b1        = (const float*)d_in[11];
    const float* w2        = (const float*)d_in[12];
    const float* b2        = (const float*)d_in[13];
    const float* ln_a_g    = (const float*)d_in[14];
    const float* ln_a_b    = (const float*)d_in[15];
    const float* ln_f_g    = (const float*)d_in[16];
    const float* ln_f_b    = (const float*)d_in[17];
    const float* gw        = (const float*)d_in[18];
    const float* gb        = (const float*)d_in[19];
    float* out = (float*)d_out;

    char* p = (char*)d_ws;
    auto carve = [&](size_t bytes) -> char* {
        char* r = p;
        p += (bytes + 255) & ~(size_t)255;
        return r;
    };
    unsigned short* wqkvBf = (unsigned short*)carve((size_t)kD3 * kD * 2);
    unsigned short* woBf   = (unsigned short*)carve((size_t)kD * kD * 2);
    unsigned short* w1Bf   = (unsigned short*)carve((size_t)kD4 * kD * 2);
    unsigned short* w2Bf   = (unsigned short*)carve((size_t)kD * kD4 * 2);
    unsigned short* gwBf   = (unsigned short*)carve((size_t)kD * kD * 2);
    float*          z      = (float*)carve((size_t)kB * kT * kD * 4);
    unsigned short* zln    = (unsigned short*)carve((size_t)kB * kT * kD * 2);
    unsigned short* qkvBf  = (unsigned short*)carve((size_t)kB * kT * kD3 * 2);
    unsigned short* oBf    = (unsigned short*)carve((size_t)kB * kT * kD * 2);
    unsigned short* h1Bf   = (unsigned short*)carve((size_t)kB * kT * kD4 * 2);
    unsigned short* candBf = (unsigned short*)carve((size_t)kB * kMem * kD * 2);
    float* memA = (float*)carve((size_t)kB * kMem * kD * 4);
    float* memB = (float*)carve((size_t)kB * kMem * kD * 4);

    cvt_bf16_kernel<<<512, 256, 0, stream>>>(wqkv, wqkvBf, kD3 * kD);
    cvt_bf16_kernel<<<512, 256, 0, stream>>>(wo, woBf, kD * kD);
    cvt_bf16_kernel<<<512, 256, 0, stream>>>(w1, w1Bf, kD4 * kD);
    cvt_bf16_kernel<<<512, 256, 0, stream>>>(w2, w2Bf, kD * kD4);
    cvt_bf16_kernel<<<512, 256, 0, stream>>>(gw, gwBf, kD * kD);

    const int rowsBT = kB * kT;  // 6144
    float* memPrev = memA;
    float* memNext = memB;

    for (int step = 0; step < kNC; ++step) {
        build_z_ln_kernel<<<rowsBT, 256, 0, stream>>>(
            in_prob, in_skill, emb_prob, emb_skill, memPrev, mem_init, pos_emb,
            ln_a_g, ln_a_b, z, zln, step);

        wmma_gemm_kernel<0><<<dim3(kD3 / 64, rowsBT / 128), 256, 0, stream>>>(
            zln, wqkvBf, bqkv, qkvBf, nullptr, nullptr, nullptr, rowsBT, kD3,
            kD);

        attn_kernel<<<kB * kH, 192, 0, stream>>>(qkvBf, oBf);

        wmma_gemm_kernel<1><<<dim3(kD / 64, rowsBT / 128), 256, 0, stream>>>(
            oBf, woBf, bo, nullptr, z, nullptr, nullptr, rowsBT, kD, kD);

        ln_kernel<<<rowsBT, 256, 0, stream>>>(z, ln_f_g, ln_f_b, zln);

        wmma_gemm_kernel<2><<<dim3(kD4 / 64, rowsBT / 128), 256, 0, stream>>>(
            zln, w1Bf, b1, h1Bf, nullptr, nullptr, nullptr, rowsBT, kD4, kD);

        wmma_gemm_kernel<1><<<dim3(kD / 64, rowsBT / 128), 256, 0, stream>>>(
            h1Bf, w2Bf, b2, nullptr, z, nullptr, nullptr, rowsBT, kD, kD4);

        finalize_kernel<<<rowsBT, 256, 0, stream>>>(z, out, candBf, step);

        if (step == 0) {
            copy_cand_kernel<<<kB * kMem, 256, 0, stream>>>(z, memNext);
        } else {
            wmma_gemm_kernel<3>
                <<<dim3(kD / 64, (kB * kMem) / 128), 256, 0, stream>>>(
                    candBf, gwBf, gb, nullptr, z, memNext, memPrev, kB * kMem,
                    kD, kD);
        }
        float* tmp = memPrev;
        memPrev = memNext;
        memNext = tmp;
    }

    hipMemcpyAsync(out + (size_t)kB * kL * kD, memPrev,
                   (size_t)kB * kMem * kD * sizeof(float),
                   hipMemcpyDeviceToDevice, stream);
}
